// GlobalAttentionNet_81243601371622
// MI455X (gfx1250) — matrix operand order repaired and checked by reading, not compiled, not run on hardware
//
#include <hip/hip_runtime.h>
#include <hip/hip_bf16.h>

// ---------------------------------------------------------------------------
// GraphSAGE(3) + GlobalAttention pool + MLP head for MI455X (gfx1250, wave32)
// GEMMs: v_wmma_f32_16x16x32_bf16, weights pre-packed into B-fragment order
// so every fragment load is contiguous (b128-pair) and wave-coalesced.
// Scatter/segment ops use global float atomics (all data lives in 192MB L2).
// ---------------------------------------------------------------------------

#define D_CH   256
#define G_CNT  64
#define L_CNT  3
#define OUT_CH 2

typedef __attribute__((ext_vector_type(16))) __bf16          v16bf;
typedef __attribute__((ext_vector_type(16))) unsigned short  v16u;
typedef __attribute__((ext_vector_type(8)))  float           v8f;

// float -> bf16 round-to-nearest-even
__device__ __forceinline__ unsigned short f2bf(float f) {
    unsigned int u = __float_as_uint(f);
    u = (u + 0x7FFFu + ((u >> 16) & 1u)) >> 16;
    return (unsigned short)u;
}

// monotone uint encoding of float for atomicMax-based segment max
__device__ __forceinline__ unsigned int fenc(float f) {
    unsigned int u = __float_as_uint(f);
    return (u & 0x80000000u) ? ~u : (u | 0x80000000u);
}
__device__ __forceinline__ float fdec(unsigned int k) {
    unsigned int u = (k & 0x80000000u) ? (k ^ 0x80000000u) : ~k;
    return __uint_as_float(u);
}

// ------------------------------- small kernels ------------------------------

__global__ void deg_kernel(const int* __restrict__ dst, float* __restrict__ deg, int E) {
    int e = blockIdx.x * blockDim.x + threadIdx.x;
    if (e < E) atomicAdd(&deg[dst[e]], 1.0f);
}

__global__ void deginv_kernel(const float* __restrict__ deg, float* __restrict__ dinv, int N) {
    int i = blockIdx.x * blockDim.x + threadIdx.x;
    if (i < N) dinv[i] = 1.0f / fmaxf(deg[i], 1.0f);
}

// one block per edge, one thread per channel: agg[dst] += h[src]
__global__ __launch_bounds__(D_CH) void scatter_kernel(
    const float* __restrict__ h, const int* __restrict__ src,
    const int* __restrict__ dst, float* __restrict__ agg, int E) {
    int e = blockIdx.x;
    if (e >= E) return;
    int s = src[e], d = dst[e], c = threadIdx.x;
    atomicAdd(&agg[(size_t)d * D_CH + c], h[(size_t)s * D_CH + c]);
}

__global__ void f2bf_kernel(const float* __restrict__ in, unsigned short* __restrict__ out, int n) {
    int i = blockIdx.x * blockDim.x + threadIdx.x;
    if (i < n) out[i] = f2bf(in[i]);
}

// bf16(in[i] * rowscale[i / D]) -- used for agg * deg_inv
__global__ void f2bf_rowscale_kernel(const float* __restrict__ in,
                                     const float* __restrict__ rowscale,
                                     unsigned short* __restrict__ out, int n) {
    int i = blockIdx.x * blockDim.x + threadIdx.x;
    if (i < n) out[i] = f2bf(in[i] * rowscale[i / D_CH]);
}

// ---------------------- weight pack: fp32 -> bf16 B-fragments ---------------
// Packed layout (KC = K/32):
//   out[ ((tn*KC + kc)*32 + lane)*16 + j ]  =  bf16( W[k * Nc + col] )
// with col = tn*16 + (lane&15),  k = kc*32 + (j>>3)*16 + (lane>>4)*8 + (j&7).
// In the GEMM, lane's B fragment = 16 contiguous shorts (32B) => b128 pair,
// and consecutive lanes read consecutive 32B chunks (fully coalesced).
__global__ void packw_kernel(const float* __restrict__ W, unsigned short* __restrict__ out,
                             int K, int Nc) {
    int i = blockIdx.x * blockDim.x + threadIdx.x;
    if (i >= K * Nc) return;
    int KC   = K >> 5;
    int j    = i & 15;
    int lane = (i >> 4) & 31;
    int grp  = i >> 9;                 // tn*KC + kc
    int tn   = grp / KC;
    int kc   = grp - tn * KC;
    int col  = (tn << 4) + (lane & 15);
    int k    = (kc << 5) + ((j >> 3) << 4) + ((lane >> 4) << 3) + (j & 7);
    out[i] = f2bf(W[(size_t)k * Nc + col]);
}

// ------------------------------- WMMA GEMM ---------------------------------
// out[M x Nc] = relu?( A@Wa (+ B@Wb) + bias ), bf16 operands, fp32 accum.
// M%16==0, K%32==0, Nc%64==0. 4 waves/block; each wave owns a 16x64 strip
// (4 N-tiles), reusing each A fragment across 4 WMMAs.
// Fragment layouts per CDNA5 ISA 7.12.2 (wave32):
//   A (16x32 bf16): lanes 0-15 row M=lane hold K 0..7 & 16..23;
//                   lanes 16-31 hold K 8..15 & 24..31.
//   B: pre-packed (see packw_kernel).
//   C/D: elem r -> row (half*8 + r), col = lane&15.
__global__ __launch_bounds__(128) void wmma_gemm_bf16(
    const unsigned short* __restrict__ A,  const unsigned short* __restrict__ WaP,
    const unsigned short* __restrict__ B,  const unsigned short* __restrict__ WbP,
    const float* __restrict__ bias, float* __restrict__ out,
    int M, int K, int Nc, int relu, int hasB) {

    const int wave    = threadIdx.x >> 5;
    const int lane    = threadIdx.x & 31;
    const int KC      = K >> 5;
    const int tilesN  = Nc >> 4;
    const int stripsN = tilesN >> 2;
    const int strips  = (M >> 4) * stripsN;
    const int strip   = blockIdx.x * 4 + wave;      // wave-uniform
    if (strip >= strips) return;

    const int tm   = strip / stripsN;
    const int tn0  = (strip - tm * stripsN) << 2;   // first of 4 N-tiles
    const int half = lane >> 4;                     // K half-select
    const int lr   = lane & 15;
    const int arow = (tm << 4) + lr;

    v8f acc[4] = {};
    for (int kc = 0; kc < KC; ++kc) {
        const int k0 = kc << 5;
        // A fragment (two contiguous 16B runs -> b128 loads)
        v16u au;
        const unsigned short* ap = A + (size_t)arow * K + k0 + (half << 3);
        #pragma unroll
        for (int j = 0; j < 8; ++j) { au[j] = ap[j]; au[8 + j] = ap[16 + j]; }
        const v16bf a = __builtin_bit_cast(v16bf, au);

        #pragma unroll
        for (int t = 0; t < 4; ++t) {
            const unsigned short* bp =
                WaP + ((((size_t)(tn0 + t) * KC + kc) << 5) + lane) * 16;
            v16u bu;
            #pragma unroll
            for (int j = 0; j < 16; ++j) bu[j] = bp[j];
            acc[t] = __builtin_amdgcn_wmma_f32_16x16x32_bf16(
                false, a, false, __builtin_bit_cast(v16bf, bu),
                (short)0, acc[t], false, false);
        }

        if (hasB) {
            v16u au2;
            const unsigned short* ap2 = B + (size_t)arow * K + k0 + (half << 3);
            #pragma unroll
            for (int j = 0; j < 8; ++j) { au2[j] = ap2[j]; au2[8 + j] = ap2[16 + j]; }
            const v16bf a2 = __builtin_bit_cast(v16bf, au2);
            #pragma unroll
            for (int t = 0; t < 4; ++t) {
                const unsigned short* bp =
                    WbP + ((((size_t)(tn0 + t) * KC + kc) << 5) + lane) * 16;
                v16u bu;
                #pragma unroll
                for (int j = 0; j < 16; ++j) bu[j] = bp[j];
                acc[t] = __builtin_amdgcn_wmma_f32_16x16x32_bf16(
                    false, a2, false, __builtin_bit_cast(v16bf, bu),
                    (short)0, acc[t], false, false);
            }
        }
    }

    #pragma unroll
    for (int t = 0; t < 4; ++t) {
        const int   bcol = ((tn0 + t) << 4) + lr;
        const float bv   = bias ? bias[bcol] : 0.0f;
        #pragma unroll
        for (int r = 0; r < 8; ++r) {
            int   orow = (tm << 4) + (half << 3) + r;
            float v    = acc[t][r] + bv;
            if (relu) v = fmaxf(v, 0.0f);
            out[(size_t)orow * Nc + bcol] = v;
        }
    }
}

// --------------------------- attention pooling -----------------------------

// one wave per node: gate[n] = dot(h[n], Wg) + bg
__global__ __launch_bounds__(256) void gate_kernel(
    const float* __restrict__ h, const float* __restrict__ Wg,
    const float* __restrict__ bg, float* __restrict__ gate, int N) {
    int wave = threadIdx.x >> 5, lane = threadIdx.x & 31;
    int n = blockIdx.x * 8 + wave;
    if (n >= N) return;
    const float* hp = h + (size_t)n * D_CH;
    float s = 0.0f;
    #pragma unroll
    for (int j = 0; j < D_CH / 32; ++j)
        s += hp[lane + 32 * j] * Wg[lane + 32 * j];
    #pragma unroll
    for (int m = 16; m >= 1; m >>= 1) s += __shfl_xor(s, m, 32);
    if (lane == 0) gate[n] = s + bg[0];
}

__global__ void segmax_kernel(const float* __restrict__ gate, const int* __restrict__ batch,
                              unsigned int* __restrict__ mkey, int N) {
    int n = blockIdx.x * blockDim.x + threadIdx.x;
    if (n < N) atomicMax(&mkey[batch[n]], fenc(gate[n]));
}

__global__ void expz_kernel(const float* __restrict__ gate, const int* __restrict__ batch,
                            const unsigned int* __restrict__ mkey,
                            float* __restrict__ eexp, float* __restrict__ z, int N) {
    int n = blockIdx.x * blockDim.x + threadIdx.x;
    if (n < N) {
        int g = batch[n];
        float e = __expf(gate[n] - fdec(mkey[g]));
        eexp[n] = e;
        atomicAdd(&z[g], e);
    }
}

// one block per node: pooled[g] += (e/z) * h[n]
__global__ __launch_bounds__(D_CH) void pooled_kernel(
    const float* __restrict__ h, const float* __restrict__ eexp,
    const float* __restrict__ z, const int* __restrict__ batch,
    float* __restrict__ pooled, int N) {
    int n = blockIdx.x;
    if (n >= N) return;
    int g = batch[n];
    float a = eexp[n] / z[g];
    int c = threadIdx.x;
    atomicAdd(&pooled[(size_t)g * D_CH + c], a * h[(size_t)n * D_CH + c]);
}

// final D->2 projection + log_softmax, one thread per graph
__global__ void head_kernel(const float* __restrict__ mlp1, const float* __restrict__ W2,
                            const float* __restrict__ b2, float* __restrict__ out, int G) {
    int g = blockIdx.x * blockDim.x + threadIdx.x;
    if (g >= G) return;
    const float* m = mlp1 + (size_t)g * D_CH;
    float v0 = b2[0], v1 = b2[1];
    for (int d = 0; d < D_CH; ++d) {
        float hv = m[d];
        v0 += hv * W2[d * OUT_CH + 0];
        v1 += hv * W2[d * OUT_CH + 1];
    }
    float mx  = fmaxf(v0, v1);
    float lse = mx + logf(__expf(v0 - mx) + __expf(v1 - mx));
    out[g * OUT_CH + 0] = v0 - lse;
    out[g * OUT_CH + 1] = v1 - lse;
}

// ------------------------------- launcher ----------------------------------

extern "C" void kernel_launch(void* const* d_in, const int* in_sizes, int n_in,
                              void* d_out, int out_size, void* d_ws, size_t ws_size,
                              hipStream_t stream) {
    (void)n_in; (void)out_size; (void)ws_size;

    const float* x     = (const float*)d_in[0];
    const int*   ei    = (const int*)  d_in[1];
    const int*   batch = (const int*)  d_in[2];
    const float* Wl    = (const float*)d_in[3];
    const float* Wr    = (const float*)d_in[4];
    const float* bias  = (const float*)d_in[5];
    const float* Wg    = (const float*)d_in[6];
    const float* bg    = (const float*)d_in[7];
    const float* W1    = (const float*)d_in[8];
    const float* b1    = (const float*)d_in[9];
    const float* W2    = (const float*)d_in[10];
    const float* b2    = (const float*)d_in[11];
    float*       out   = (float*)d_out;

    const int N = in_sizes[2];        // 10000 (multiple of 16)
    const int E = in_sizes[1] / 2;    // 320000
    const int* src = ei;
    const int* dst = ei + E;

    // workspace carve (256B aligned)
    char*  ws  = (char*)d_ws;
    size_t off = 0;
    auto alloc = [&](size_t bytes) -> void* {
        off = (off + 255) & ~(size_t)255;
        void* p = ws + off;
        off += bytes;
        return p;
    };
    const size_t ND = (size_t)N * D_CH;
    float*          deg    = (float*)         alloc((size_t)N * 4);
    float*          dinv   = (float*)         alloc((size_t)N * 4);
    float*          hA     = (float*)         alloc(ND * 4);
    float*          hB     = (float*)         alloc(ND * 4);
    float*          agg    = (float*)         alloc(ND * 4);
    unsigned short* hbf    = (unsigned short*)alloc(ND * 2);
    unsigned short* abf    = (unsigned short*)alloc(ND * 2);
    unsigned short* wabf   = (unsigned short*)alloc((size_t)D_CH * D_CH * 2);
    unsigned short* wbbf   = (unsigned short*)alloc((size_t)D_CH * D_CH * 2);
    float*          gate   = (float*)         alloc((size_t)N * 4);
    float*          eexp   = (float*)         alloc((size_t)N * 4);
    unsigned int*   mkey   = (unsigned int*)  alloc((size_t)G_CNT * 4);
    float*          z      = (float*)         alloc((size_t)G_CNT * 4);
    float*          pooled = (float*)         alloc((size_t)G_CNT * D_CH * 4);
    unsigned short* pbf    = (unsigned short*)alloc((size_t)G_CNT * D_CH * 2);
    unsigned short* w1bf   = (unsigned short*)alloc((size_t)D_CH * D_CH * 2);
    float*          mlp1   = (float*)         alloc((size_t)G_CNT * D_CH * 4);

    // degree and 1/max(deg,1)
    hipMemsetAsync(deg, 0, (size_t)N * 4, stream);
    deg_kernel<<<(E + 255) / 256, 256, 0, stream>>>(dst, deg, E);
    deginv_kernel<<<(N + 255) / 256, 256, 0, stream>>>(deg, dinv, N);

    const int wBlocks = (D_CH * D_CH + 255) / 256;

    // SAGE layers
    const float* hin  = x;
    float*       hout = hA;
    for (int l = 0; l < L_CNT; ++l) {
        hipMemsetAsync(agg, 0, ND * 4, stream);
        scatter_kernel<<<E, D_CH, 0, stream>>>(hin, src, dst, agg, E);

        int ndBlocks = (int)((ND + 255) / 256);
        f2bf_rowscale_kernel<<<ndBlocks, 256, 0, stream>>>(agg, dinv, abf, (int)ND);
        f2bf_kernel<<<ndBlocks, 256, 0, stream>>>(hin, hbf, (int)ND);
        packw_kernel<<<wBlocks, 256, 0, stream>>>(Wl + (size_t)l * D_CH * D_CH, wabf, D_CH, D_CH);
        packw_kernel<<<wBlocks, 256, 0, stream>>>(Wr + (size_t)l * D_CH * D_CH, wbbf, D_CH, D_CH);

        int strips = (N >> 4) * ((D_CH >> 4) >> 2);
        wmma_gemm_bf16<<<(strips + 3) / 4, 128, 0, stream>>>(
            abf, wabf, hbf, wbbf, bias + (size_t)l * D_CH, hout,
            N, D_CH, D_CH, /*relu=*/1, /*hasB=*/1);

        hin  = hout;
        hout = (hout == hA) ? hB : hA;
    }
    const float* h = hin; // final node features [N, D]

    // global attention pooling
    gate_kernel<<<(N + 7) / 8, 256, 0, stream>>>(h, Wg, bg, gate, N);
    hipMemsetAsync(mkey, 0, (size_t)G_CNT * 4, stream);
    segmax_kernel<<<(N + 255) / 256, 256, 0, stream>>>(gate, batch, mkey, N);
    hipMemsetAsync(z, 0, (size_t)G_CNT * 4, stream);
    expz_kernel<<<(N + 255) / 256, 256, 0, stream>>>(gate, batch, mkey, eexp, z, N);
    hipMemsetAsync(pooled, 0, (size_t)G_CNT * D_CH * 4, stream);
    pooled_kernel<<<N, D_CH, 0, stream>>>(h, eexp, z, batch, pooled, N);

    // head: relu(pooled @ W1 + b1) via WMMA, then D->2 + log_softmax
    f2bf_kernel<<<(G_CNT * D_CH + 255) / 256, 256, 0, stream>>>(pooled, pbf, G_CNT * D_CH);
    packw_kernel<<<wBlocks, 256, 0, stream>>>(W1, w1bf, D_CH, D_CH);
    int strips2 = (G_CNT >> 4) * ((D_CH >> 4) >> 2);
    wmma_gemm_bf16<<<(strips2 + 3) / 4, 128, 0, stream>>>(
        pbf, w1bf, nullptr, nullptr, b1, mlp1,
        G_CNT, D_CH, D_CH, /*relu=*/1, /*hasB=*/0);

    head_kernel<<<(G_CNT + 63) / 64, 64, 0, stream>>>(mlp1, W2, b2, out, G_CNT);
}
